// SelfAttention_33956011442194
// MI455X (gfx1250) — matrix operand order repaired and checked by reading, not compiled, not run on hardware
//
#include <hip/hip_runtime.h>
#include <math.h>

#define SEQ 4096
#define EMB 2048

#define BM 128
#define BN 128
#define BK 64
#define LDP (BK + 8)   // padded LDS pitch (72 bf16 = 36 dwords) to avoid bank conflicts

typedef __attribute__((ext_vector_type(16))) __bf16 v16bf;
typedef __attribute__((ext_vector_type(8)))  float  v8f;
typedef __attribute__((ext_vector_type(4)))  float  v4f;
typedef __attribute__((ext_vector_type(4)))  int    v4i;

#define AS1 __attribute__((address_space(1)))
#define AS3 __attribute__((address_space(3)))

#ifndef __has_builtin
#define __has_builtin(x) 0
#endif

#if __has_builtin(__builtin_amdgcn_global_load_async_to_lds_b128)
#define HAVE_ASYNC_LDS 1
#else
#define HAVE_ASYNC_LDS 0
#endif

union Frag {
  v16bf v;
  uint4 u[2];
};

// Scheduling fence: keep fragment-load groups and WMMA groups from being
// interleaved back into a load->wait->wmma chain by the pre-RA scheduler.
__device__ __forceinline__ void sched_fence() {
#if __has_builtin(__builtin_amdgcn_sched_barrier)
  __builtin_amdgcn_sched_barrier(0);
#endif
}

// ---------------------------------------------------------------- staging
// Copy a ROWS x BK bf16 tile (row-major, leading dim ldk) into LDS (pitch LDP)
// via the CDNA5 async global->LDS DMA path (ASYNCcnt). Fully unrolled:
// ROWS*8 16-byte chunks spread over 256 threads.
template <int ROWS>
__device__ __forceinline__ void stage_tile(const __bf16* __restrict__ gbase, size_t ldk,
                                           __bf16 (*lds)[LDP], int tid) {
  constexpr int TOTAL = ROWS * (BK / 8);   // 16B chunks
  constexpr int ITERS = TOTAL / 256;
#pragma unroll
  for (int i = 0; i < ITERS; ++i) {
    const int c  = tid + i * 256;
    const int r  = c >> 3;
    const int c8 = c & 7;
    const __bf16* g = gbase + (size_t)r * ldk + c8 * 8;
    __bf16* l = &lds[r][c8 * 8];
#if HAVE_ASYNC_LDS
    __builtin_amdgcn_global_load_async_to_lds_b128((AS1 v4i*)g, (AS3 v4i*)l, 0, 0);
#else
    *(uint4*)l = *(const uint4*)g;
#endif
  }
}

__device__ __forceinline__ void wait_async_tiles() {
#if HAVE_ASYNC_LDS
#if __has_builtin(__builtin_amdgcn_s_wait_asynccnt)
  __builtin_amdgcn_s_wait_asynccnt(0);
#else
  asm volatile("s_wait_asynccnt 0" ::: "memory");
#endif
#endif
}

// ---------------------------------------------------------------- GEMM core
// out[m][n] = sum_k A[m][k] * B[n][k]   (A: MxK bf16, B: NxK bf16, both row-major)
// MODE 0: store bf16 row-major           -> outb[m*N + n]
// MODE 1: store bf16 transposed          -> outb[n*M + m]        (V^T)
// MODE 2: store f32 * scale              -> outf[m*N + n]        (scores)
// MODE 3: store f32 * linv[m]            -> outf[m*N + n]        (p@v, folded softmax norm)
//
// 8 waves in a 4(M) x 2(N) grid; each wave owns a 32x64 strip:
// 2 M-strips x 4 N-tiles = 8 accumulators, 16 WMMA per BK=64 tile.
template <int MODE>
__global__ __launch_bounds__(256) void gemm_core(const __bf16* __restrict__ A,
                                                 const __bf16* __restrict__ Bm,
                                                 __bf16* __restrict__ outb,
                                                 float* __restrict__ outf,
                                                 const float* __restrict__ linv,
                                                 int M, int N, int K, float scale) {
  __shared__ __bf16 As[2][BM][LDP];
  __shared__ __bf16 Bs[2][BN][LDP];

  const int tid  = threadIdx.x;
  const int lane = tid & 31;
  const int wave = tid >> 5;
  const int h    = lane >> 4;
  const int l16  = lane & 15;
  const int mw   = (wave >> 1) * 32;          // wave's 32-row strip in block tile
  const int nw   = (wave & 1) * 64;           // wave's 64-col strip in block tile
  const int m_base = blockIdx.y * BM;
  const int n_base = blockIdx.x * BN;

  const __bf16* Ablk = A  + (size_t)m_base * K;
  const __bf16* Bblk = Bm + (size_t)n_base * K;

  v8f acc[2][4] = {{{}, {}, {}, {}}, {{}, {}, {}, {}}};

  const int nkb = K / BK;

  // prologue: stage tile 0
  stage_tile<BM>(Ablk, K, As[0], tid);
  stage_tile<BN>(Bblk, K, Bs[0], tid);
  wait_async_tiles();
  __syncthreads();

  for (int kb = 0; kb < nkb; ++kb) {
    const int cur = kb & 1;
    const int nxt = cur ^ 1;
    const bool have_next = (kb + 1) < nkb;

    // issue the next tile's global->LDS DMA before computing on the current one
    if (have_next) {
      const int k0 = (kb + 1) * BK;
      stage_tile<BM>(Ablk + k0, K, As[nxt], tid);
      stage_tile<BN>(Bblk + k0, K, Bs[nxt], tid);
    }

    // ---- compute on current LDS tile: 2 K-steps of 32 ----
#pragma unroll
    for (int s = 0; s < 2; ++s) {
      Frag a[2];
      Frag b[4];
#pragma unroll
      for (int i = 0; i < 2; ++i) {
        const __bf16* ap = &As[cur][mw + i * 16 + l16][s * 32 + h * 8];
        a[i].u[0] = *(const uint4*)ap;
        a[i].u[1] = *(const uint4*)(ap + 16);
      }
#pragma unroll
      for (int t = 0; t < 4; ++t) {
        const __bf16* bp = &Bs[cur][nw + t * 16 + l16][s * 32 + h * 8];
        b[t].u[0] = *(const uint4*)bp;
        b[t].u[1] = *(const uint4*)(bp + 16);
      }
      sched_fence();   // loads above, WMMA burst below (one dscnt wait)
#pragma unroll
      for (int t = 0; t < 4; ++t) {
#pragma unroll
        for (int i = 0; i < 2; ++i) {
          acc[i][t] = __builtin_amdgcn_wmma_f32_16x16x32_bf16(
              false, a[i].v, false, b[t].v, (short)0, acc[i][t], false, false);
        }
      }
    }

    wait_async_tiles();
    __syncthreads();
  }

  // C/D layout: VGPR r, lanes 0-15 -> M=r, lanes 16-31 -> M=8+r, N = lane&15
#pragma unroll
  for (int i = 0; i < 2; ++i) {
#pragma unroll
    for (int t = 0; t < 4; ++t) {
      const int n = n_base + nw + t * 16 + l16;
#pragma unroll
      for (int r = 0; r < 8; ++r) {
        const int m = m_base + mw + i * 16 + h * 8 + r;
        const float val = acc[i][t][r];
        if (MODE == 0) outb[(size_t)m * N + n] = (__bf16)val;
        if (MODE == 1) outb[(size_t)n * M + m] = (__bf16)val;
        if (MODE == 2) outf[(size_t)m * N + n] = val * scale;
        if (MODE == 3) outf[(size_t)m * N + n] = val * linv[m];
      }
    }
  }
}

// ---------------------------------------------------------------- convert
typedef struct { __bf16 x, y, z, w; } bf16x4;

__global__ __launch_bounds__(256) void cvt_f32_bf16_v4(const v4f* __restrict__ src,
                                                       bf16x4* __restrict__ dst, int n4) {
  for (int i = blockIdx.x * blockDim.x + threadIdx.x; i < n4; i += gridDim.x * blockDim.x) {
    v4f s = src[i];
    bf16x4 d;
    d.x = (__bf16)s[0]; d.y = (__bf16)s[1]; d.z = (__bf16)s[2]; d.w = (__bf16)s[3];
    dst[i] = d;
  }
}

// ---------------------------------------------------------------- softmax
// One block per row. Writes UNNORMALIZED exp(s - max) as bf16 and 1/rowsum
// (normalization is folded into the final GEMM's store).
__global__ __launch_bounds__(256) void softmax_rows(const float* __restrict__ S_,
                                                    __bf16* __restrict__ P,
                                                    float* __restrict__ linv, int N) {
  const int row = blockIdx.x;
  const int tid = threadIdx.x;
  const float* sr = S_ + (size_t)row * N;
  __shared__ float red[256];

  float mx = -INFINITY;
  for (int j = tid; j < N; j += 256) mx = fmaxf(mx, sr[j]);
  red[tid] = mx;
  __syncthreads();
  for (int s = 128; s > 0; s >>= 1) {
    if (tid < s) red[tid] = fmaxf(red[tid], red[tid + s]);
    __syncthreads();
  }
  mx = red[0];
  __syncthreads();

  float sum = 0.0f;
  for (int j = tid; j < N; j += 256) {
    float e = __expf(sr[j] - mx);
    P[(size_t)row * N + j] = (__bf16)e;
    sum += e;
  }
  red[tid] = sum;
  __syncthreads();
  for (int s = 128; s > 0; s >>= 1) {
    if (tid < s) red[tid] += red[tid + s];
    __syncthreads();
  }
  if (tid == 0) linv[row] = 1.0f / red[0];
}

// ---------------------------------------------------------------- launch
extern "C" void kernel_launch(void* const* d_in, const int* in_sizes, int n_in,
                              void* d_out, int out_size, void* d_ws, size_t ws_size,
                              hipStream_t stream) {
  const float* x  = (const float*)d_in[0];
  const float* Wq = (const float*)d_in[1];
  const float* Wk = (const float*)d_in[2];
  const float* Wv = (const float*)d_in[3];
  float* out = (float*)d_out;

  char* ws = (char*)d_ws;
  size_t off = 0;
  __bf16* xb  = (__bf16*)(ws + off); off += (size_t)SEQ * EMB * 2;   // 16 MB
  __bf16* wqb = (__bf16*)(ws + off); off += (size_t)EMB * EMB * 2;   //  8 MB
  __bf16* wkb = (__bf16*)(ws + off); off += (size_t)EMB * EMB * 2;   //  8 MB
  __bf16* wvb = (__bf16*)(ws + off); off += (size_t)EMB * EMB * 2;   //  8 MB
  __bf16* qb  = (__bf16*)(ws + off); off += (size_t)SEQ * EMB * 2;   // 16 MB
  __bf16* kb  = (__bf16*)(ws + off); off += (size_t)SEQ * EMB * 2;   // 16 MB
  __bf16* vtb = (__bf16*)(ws + off); off += (size_t)EMB * SEQ * 2;   // 16 MB (V^T [E][S])
  float*  sb  = (float* )(ws + off); off += (size_t)SEQ * SEQ * 4;   // 64 MB
  __bf16* pb  = (__bf16*)(ws + off); off += (size_t)SEQ * SEQ * 2;   // 32 MB
  float*  li  = (float* )(ws + off); off += (size_t)SEQ * 4;         // 16 KB

  // 1. fp32 -> bf16 conversions (vectorized: 16B in, 8B out per thread-iter)
  cvt_f32_bf16_v4<<<2048, 256, 0, stream>>>((const v4f*)x,  (bf16x4*)xb,  SEQ * EMB / 4);
  cvt_f32_bf16_v4<<<1024, 256, 0, stream>>>((const v4f*)Wq, (bf16x4*)wqb, EMB * EMB / 4);
  cvt_f32_bf16_v4<<<1024, 256, 0, stream>>>((const v4f*)Wk, (bf16x4*)wkb, EMB * EMB / 4);
  cvt_f32_bf16_v4<<<1024, 256, 0, stream>>>((const v4f*)Wv, (bf16x4*)wvb, EMB * EMB / 4);

  const dim3 blk(256);

  // 2. projections: q, k (row-major), v stored transposed
  {
    dim3 grid(EMB / BN, SEQ / BM);  // 16 x 32
    gemm_core<0><<<grid, blk, 0, stream>>>(xb, wqb, qb,  nullptr, nullptr, SEQ, EMB, EMB, 1.0f);
    gemm_core<0><<<grid, blk, 0, stream>>>(xb, wkb, kb,  nullptr, nullptr, SEQ, EMB, EMB, 1.0f);
    gemm_core<1><<<grid, blk, 0, stream>>>(xb, wvb, vtb, nullptr, nullptr, SEQ, EMB, EMB, 1.0f);
  }

  // 3. scores s = q k^T / sqrt(E)  (fp32, mostly L2-resident)
  {
    dim3 grid(SEQ / BN, SEQ / BM);  // 32 x 32
    const float scale = 1.0f / sqrtf((float)EMB);
    gemm_core<2><<<grid, blk, 0, stream>>>(qb, kb, nullptr, sb, nullptr, SEQ, SEQ, EMB, scale);
  }

  // 4. row softmax -> unnormalized p (bf16) + 1/rowsum
  softmax_rows<<<SEQ, 256, 0, stream>>>(sb, pb, li, SEQ);

  // 5. out = (p @ v) * (1/rowsum)
  {
    dim3 grid(EMB / BN, SEQ / BM);  // 16 x 32
    gemm_core<3><<<grid, blk, 0, stream>>>(pb, vtb, nullptr, out, li, SEQ, EMB, SEQ, 1.0f);
  }
}
